// FeaturePropagation_80083960201799
// MI455X (gfx1250) — matrix operand order repaired and checked by reading, not compiled, not run on hardware
//
#include <hip/hip_runtime.h>
#include <hip/hip_bf16.h>

typedef __attribute__((ext_vector_type(16))) __bf16 v16bf;
typedef __attribute__((ext_vector_type(8)))  __bf16 v8bf;
typedef __attribute__((ext_vector_type(8)))  float  v8f;

#define BB    4
#define NN    16384
#define SSP   4096
#define D1C   128
#define D2C   256
#define INCH  384
#define OUT0  256
#define OUT1  128
#define MROWS (BB * NN)   // 65536 rows through the MLP

#if defined(__gfx1250__) && __has_builtin(__builtin_amdgcn_global_load_async_to_lds_b128)
#define HAVE_ASYNC_LDS 1
typedef int v4i __attribute__((vector_size(16)));          // pointee type the builtin expects
typedef __attribute__((address_space(1))) v4i gbl_v4i_t;   // prints as "__device__" AS
typedef __attribute__((address_space(3))) v4i lds_v4i_t;   // prints as "__shared__" AS
#endif

// ---------------------------------------------------------------------------
// small utility kernels
// ---------------------------------------------------------------------------
__global__ __launch_bounds__(256) void zero_f32(float* p, int n) {
    int i = blockIdx.x * 256 + threadIdx.x;
    if (i < n) p[i] = 0.0f;
}

__global__ __launch_bounds__(256) void cvt_f32_bf16(const float* __restrict__ src,
                                                    __bf16* __restrict__ dst, int n) {
    int i = blockIdx.x * 256 + threadIdx.x;
    if (i < n) dst[i] = (__bf16)src[i];
}

// ---------------------------------------------------------------------------
// 3-NN search: one thread per query point; xyz2 staged through LDS in 12KB
// tiles using GLOBAL_LOAD_ASYNC_TO_LDS (ASYNCcnt) when available.
// Tile is stored contiguously: point j at stile[3j..3j+2]; stride-3 dword
// access across lanes is bank-conflict free (gcd(3,64)=1).
// ---------------------------------------------------------------------------
__global__ __launch_bounds__(256) void knn3_kernel(const float* __restrict__ xyz1,
                                                   const float* __restrict__ xyz2,
                                                   int* __restrict__ idx,
                                                   float* __restrict__ wgt) {
    __shared__ float stile[3072];                   // 1024 points * 3 floats
    const int q = blockIdx.x * 256 + threadIdx.x;   // global query over B*N
    const int b = q >> 14;                          // N = 16384
    const float qx = xyz1[(size_t)q * 3 + 0];
    const float qy = xyz1[(size_t)q * 3 + 1];
    const float qz = xyz1[(size_t)q * 3 + 2];
    const float* base2 = xyz2 + (size_t)b * SSP * 3;

    float d0 = 3.4e38f, d1 = 3.4e38f, d2 = 3.4e38f;
    int   i0 = 0, i1 = 0, i2 = 0;

    for (int s0 = 0; s0 < SSP; s0 += 1024) {
        const float* gsrc = base2 + (size_t)s0 * 3;   // 3072 contiguous floats
#if defined(HAVE_ASYNC_LDS)
        // 3072 floats = 768 x b128; 256 threads x 3 async ops each
        {
            const float* gp = gsrc + threadIdx.x * 4;
            float*       lp = stile + threadIdx.x * 4;
            __builtin_amdgcn_global_load_async_to_lds_b128(
                (gbl_v4i_t*)(gp),        (lds_v4i_t*)(lp),        0, 0);
            __builtin_amdgcn_global_load_async_to_lds_b128(
                (gbl_v4i_t*)(gp + 1024), (lds_v4i_t*)(lp + 1024), 0, 0);
            __builtin_amdgcn_global_load_async_to_lds_b128(
                (gbl_v4i_t*)(gp + 2048), (lds_v4i_t*)(lp + 2048), 0, 0);
        }
#if __has_builtin(__builtin_amdgcn_s_wait_asynccnt)
        __builtin_amdgcn_s_wait_asynccnt(0);
#else
        asm volatile("s_wait_asynccnt 0" ::: "memory");
#endif
#else
        for (int t = threadIdx.x; t < 3072; t += 256) stile[t] = gsrc[t];
#endif
        __syncthreads();
        for (int j = 0; j < 1024; ++j) {
            float dx = qx - stile[j * 3 + 0];
            float dy = qy - stile[j * 3 + 1];
            float dz = qz - stile[j * 3 + 2];
            float d = dx * dx + dy * dy + dz * dz;
            int   si = s0 + j;
            if (d < d2) {
                if (d < d1) {
                    if (d < d0) { d2 = d1; i2 = i1; d1 = d0; i1 = i0; d0 = d; i0 = si; }
                    else        { d2 = d1; i2 = i1; d1 = d;  i1 = si; }
                } else          { d2 = d;  i2 = si; }
            }
        }
        __syncthreads();
    }
    float r0 = 1.0f / (d0 + 1e-8f);
    float r1 = 1.0f / (d1 + 1e-8f);
    float r2 = 1.0f / (d2 + 1e-8f);
    float rs = 1.0f / (r0 + r1 + r2);
    idx[(size_t)q * 3 + 0] = i0;  wgt[(size_t)q * 3 + 0] = r0 * rs;
    idx[(size_t)q * 3 + 1] = i1;  wgt[(size_t)q * 3 + 1] = r1 * rs;
    idx[(size_t)q * 3 + 2] = i2;  wgt[(size_t)q * 3 + 2] = r2 * rs;
}

// ---------------------------------------------------------------------------
// build x0 = concat(points1, interpolate(points2)) as bf16, (B*N, 384)
// one block per query row, one thread per channel
// ---------------------------------------------------------------------------
__global__ __launch_bounds__(384) void build_x0_kernel(const float* __restrict__ points1,
                                                       const float* __restrict__ points2,
                                                       const int* __restrict__ idx,
                                                       const float* __restrict__ wgt,
                                                       __bf16* __restrict__ x0) {
    const int q = blockIdx.x;          // 0..MROWS-1
    const int c = threadIdx.x;         // 0..383
    const int b = q >> 14;
    float v;
    if (c < D1C) {
        v = points1[(size_t)q * D1C + c];
    } else {
        const int cc = c - D1C;
        const float* p2 = points2 + (size_t)b * SSP * D2C;
        const int i0 = idx[(size_t)q * 3 + 0];
        const int i1 = idx[(size_t)q * 3 + 1];
        const int i2 = idx[(size_t)q * 3 + 2];
        const float w0 = wgt[(size_t)q * 3 + 0];
        const float w1 = wgt[(size_t)q * 3 + 1];
        const float w2 = wgt[(size_t)q * 3 + 2];
        v = w0 * p2[(size_t)i0 * D2C + cc]
          + w1 * p2[(size_t)i1 * D2C + cc]
          + w2 * p2[(size_t)i2 * D2C + cc];
    }
    x0[(size_t)q * INCH + c] = (__bf16)v;
}

// ---------------------------------------------------------------------------
// bf16 WMMA GEMM:  Y[M x N] = X[M x K] * W[N x K]^T + bias   (f32 out)
// block = 256 threads = 8 waves; wave computes a 16x64 tile (4 wmma accumulators)
// grid = (M/128, N/64)
// ---------------------------------------------------------------------------
template <int K>
__global__ __launch_bounds__(256)
void gemm_bf16_wmma(const __bf16* __restrict__ X, const __bf16* __restrict__ W,
                    const float* __restrict__ bias, float* __restrict__ Y, int N) {
    const int lane = threadIdx.x & 31;
    const int wave = threadIdx.x >> 5;
    const int kh   = lane >> 4;        // lane half selects K sub-groups
    const int r    = lane & 15;
    const int row0 = blockIdx.x * 128 + wave * 16;
    const int col0 = blockIdx.y * 64;

    // A row for this lane (16 contiguous K per k-step, K = 16*kh + [0..15])
    const __bf16* xrow = X + (size_t)(row0 + r) * K + kh * 16;

    v8f acc[4] = {};

    for (int k0 = 0; k0 < K; k0 += 32) {
        __builtin_prefetch(xrow + k0 + 128, 0, 1);   // global_prefetch_b8
        v8bf alo = *(const v8bf*)(xrow + k0);        // K = k0 + 16*kh + [0..7]
        v8bf ahi = *(const v8bf*)(xrow + k0 + 8);    // K = k0 + 16*kh + 8 + [0..7]
        v16bf a = __builtin_shufflevector(alo, ahi,
                                          0, 1, 2, 3, 4, 5, 6, 7,
                                          8, 9, 10, 11, 12, 13, 14, 15);
#pragma unroll
        for (int t = 0; t < 4; ++t) {
            const int col = col0 + t * 16 + r;       // B lane holds column (lane&15)
            v16bf bf = *(const v16bf*)(W + (size_t)col * K + k0 + kh * 16);
            acc[t] = __builtin_amdgcn_wmma_f32_16x16x32_bf16(
                false, a, false, bf, (short)0, acc[t], false, false);
        }
    }

#pragma unroll
    for (int t = 0; t < 4; ++t) {
        const int col = col0 + t * 16 + r;
        const float bv = bias[col];
#pragma unroll
        for (int p = 0; p < 8; ++p) {
            const int row = row0 + p + 8 * kh;       // C/D layout: M = p + 8*(lane>=16)
            Y[(size_t)row * N + col] = acc[t][p] + bv;
        }
    }
}

// ---------------------------------------------------------------------------
// BatchNorm: per-channel sum / sum-of-squares partials + global f32 atomics
// block covers 65536 flat elements; C divides 256 so each thread's channel is fixed
// ---------------------------------------------------------------------------
__global__ __launch_bounds__(256)
void bn_stats(const float* __restrict__ Y, float* __restrict__ sum,
              float* __restrict__ ssq, int C) {
    const size_t start = (size_t)blockIdx.x * 65536;
    const int c = (int)(threadIdx.x & (C - 1));
    float s = 0.0f, q = 0.0f;
    for (int i = 0; i < 256; ++i) {
        float v = Y[start + threadIdx.x + (size_t)i * 256];
        s += v;
        q += v * v;
    }
    atomicAdd(&sum[c], s);
    atomicAdd(&ssq[c], q);
}

__global__ void bn_finalize(const float* __restrict__ sum, const float* __restrict__ ssq,
                            const float* __restrict__ gamma, const float* __restrict__ beta,
                            float* __restrict__ scale, float* __restrict__ shift,
                            int C, float invM) {
    const int c = threadIdx.x;
    if (c < C) {
        float m  = sum[c] * invM;
        float vr = ssq[c] * invM - m * m;
        float sc = gamma[c] * rsqrtf(vr + 1e-5f);
        scale[c] = sc;
        shift[c] = beta[c] - m * sc;
    }
}

// normalize + ReLU + convert to bf16 (feeds next GEMM's A operand)
__global__ __launch_bounds__(256)
void bn_apply_bf16(const float* __restrict__ Y, const float* __restrict__ scale,
                   const float* __restrict__ shift, __bf16* __restrict__ X,
                   size_t n, int C) {
    size_t i = (size_t)blockIdx.x * 256 + threadIdx.x;
    if (i < n) {
        int c = (int)(i & (size_t)(C - 1));
        float v = Y[i] * scale[c] + shift[c];
        X[i] = (__bf16)fmaxf(v, 0.0f);
    }
}

// normalize + ReLU in place (final output)
__global__ __launch_bounds__(256)
void bn_apply_f32(float* __restrict__ Y, const float* __restrict__ scale,
                  const float* __restrict__ shift, size_t n, int C) {
    size_t i = (size_t)blockIdx.x * 256 + threadIdx.x;
    if (i < n) {
        int c = (int)(i & (size_t)(C - 1));
        float v = Y[i] * scale[c] + shift[c];
        Y[i] = fmaxf(v, 0.0f);
    }
}

// ---------------------------------------------------------------------------
extern "C" void kernel_launch(void* const* d_in, const int* in_sizes, int n_in,
                              void* d_out, int out_size, void* d_ws, size_t ws_size,
                              hipStream_t stream) {
    const float* xyz1    = (const float*)d_in[0];
    const float* xyz2    = (const float*)d_in[1];
    const float* points1 = (const float*)d_in[2];
    const float* points2 = (const float*)d_in[3];
    const float* w0  = (const float*)d_in[4];
    const float* b0  = (const float*)d_in[5];
    const float* g0  = (const float*)d_in[6];
    const float* be0 = (const float*)d_in[7];
    const float* w1  = (const float*)d_in[8];
    const float* b1  = (const float*)d_in[9];
    const float* g1  = (const float*)d_in[10];
    const float* be1 = (const float*)d_in[11];
    float* out = (float*)d_out;

    char* ws = (char*)d_ws;
    size_t off = 0;
    auto carve = [&](size_t bytes) -> void* {
        void* p = ws + off;
        off = (off + bytes + 255) & ~(size_t)255;
        return p;
    };

    int*    idx  = (int*)carve((size_t)MROWS * 3 * sizeof(int));
    float*  wgt  = (float*)carve((size_t)MROWS * 3 * sizeof(float));
    __bf16* Wb0  = (__bf16*)carve((size_t)OUT0 * INCH * sizeof(__bf16));
    __bf16* Wb1  = (__bf16*)carve((size_t)OUT1 * OUT0 * sizeof(__bf16));
    float*  st   = (float*)carve(1536 * sizeof(float));   // sums/ssq/scale/shift, both layers
    __bf16* x0   = (__bf16*)carve((size_t)MROWS * INCH * sizeof(__bf16));
    float*  y0   = (float*)carve((size_t)MROWS * OUT0 * sizeof(float));
    __bf16* x1   = (__bf16*)carve((size_t)MROWS * OUT0 * sizeof(__bf16));

    float* sum0 = st;          // 256
    float* ssq0 = st + 256;    // 256
    float* sc0  = st + 512;    // 256
    float* sh0  = st + 768;    // 256
    float* sum1 = st + 1024;   // 128
    float* ssq1 = st + 1152;   // 128
    float* sc1  = st + 1280;   // 128
    float* sh1  = st + 1408;   // 128

    // stats accumulators must start at zero every call (atomics)
    zero_f32<<<(1536 + 255) / 256, 256, 0, stream>>>(st, 1536);

    // weights to bf16
    cvt_f32_bf16<<<(OUT0 * INCH + 255) / 256, 256, 0, stream>>>(w0, Wb0, OUT0 * INCH);
    cvt_f32_bf16<<<(OUT1 * OUT0 + 255) / 256, 256, 0, stream>>>(w1, Wb1, OUT1 * OUT0);

    // 3-NN + interpolation weights
    knn3_kernel<<<MROWS / 256, 256, 0, stream>>>(xyz1, xyz2, idx, wgt);

    // concat(points1, interpolated) -> bf16
    build_x0_kernel<<<MROWS, 384, 0, stream>>>(points1, points2, idx, wgt, x0);

    // layer 0: GEMM (65536x384) x (384->256)
    gemm_bf16_wmma<INCH><<<dim3(MROWS / 128, OUT0 / 64), 256, 0, stream>>>(x0, Wb0, b0, y0, OUT0);
    bn_stats<<<(int)(((size_t)MROWS * OUT0) / 65536), 256, 0, stream>>>(y0, sum0, ssq0, OUT0);
    bn_finalize<<<1, OUT0, 0, stream>>>(sum0, ssq0, g0, be0, sc0, sh0, OUT0, 1.0f / (float)MROWS);
    {
        size_t n = (size_t)MROWS * OUT0;
        bn_apply_bf16<<<(int)((n + 255) / 256), 256, 0, stream>>>(y0, sc0, sh0, x1, n, OUT0);
    }

    // layer 1: GEMM (65536x256) x (256->128) directly into d_out
    gemm_bf16_wmma<OUT0><<<dim3(MROWS / 128, OUT1 / 64), 256, 0, stream>>>(x1, Wb1, b1, out, OUT1);
    bn_stats<<<(int)(((size_t)MROWS * OUT1) / 65536), 256, 0, stream>>>(out, sum1, ssq1, OUT1);
    bn_finalize<<<1, OUT1, 0, stream>>>(sum1, ssq1, g1, be1, sc1, sh1, OUT1, 1.0f / (float)MROWS);
    {
        size_t n = (size_t)MROWS * OUT1;
        bn_apply_f32<<<(int)((n + 255) / 256), 256, 0, stream>>>(out, sc1, sh1, n, OUT1);
    }
}
